// MultiHeadedSelfAttentionModule_83811991814370
// MI455X (gfx1250) — matrix operand order repaired and checked by reading, not compiled, not run on hardware
//
#include <hip/hip_runtime.h>

// ---------------------------------------------------------------------------
// Transformer-XL relative MHSA for gfx1250 (MI455X).
// - All matmuls via v_wmma_f32_16x16x32_bf16 (wave32, fp32 accumulate).
// - GEMM tiles staged with GLOBAL_LOAD_ASYNC_TO_LDS_B128 (ASYNCcnt path),
//   double-buffered so the async fill of K-step i+1 overlaps WMMAs of step i.
// B=4, S=2048, D=512, H=8, dh=64.
// ---------------------------------------------------------------------------

typedef __bf16 bf16;
typedef __attribute__((ext_vector_type(8)))  __bf16 bf16x8;
typedef __attribute__((ext_vector_type(16))) __bf16 bf16x16;
typedef __attribute__((ext_vector_type(8)))  float  fx8;
typedef __attribute__((ext_vector_type(4)))  float  fx4;

static constexpr int kB  = 4;
static constexpr int kS  = 2048;
static constexpr int kD  = 512;
static constexpr int kH  = 8;

__device__ __forceinline__ fx8 zero8() {
  fx8 z;
#pragma unroll
  for (int i = 0; i < 8; ++i) z[i] = 0.0f;
  return z;
}

// Async global->LDS 128-bit copy (per enabled lane). LDS byte offset is the
// low 32 bits of the generic shared-aperture address (ISA 10.2).
__device__ __forceinline__ void async_b128(const bf16* gsrc, bf16* ldst) {
  unsigned lds = (unsigned)(unsigned long long)ldst;
  asm volatile("global_load_async_to_lds_b128 %0, %1, off"
               :: "v"(lds), "v"(gsrc) : "memory");
}
__device__ __forceinline__ void wait_async0() {
  asm volatile("s_wait_asynccnt 0" ::: "memory");
}

// A-fragment (16x32 bf16, 16-bit A layout): lane l -> row m=l&15,
// k = kb..kb+7 and kb+16..kb+23 with kb = (l>=16)*8. Row-major tile, ld=32.
__device__ __forceinline__ bf16x16 load_frag_a(const bf16* rowbase, int lane) {
  const int kb = (lane >> 4) << 3;
  bf16x8 lo = *reinterpret_cast<const bf16x8*>(rowbase + kb);
  bf16x8 hi = *reinterpret_cast<const bf16x8*>(rowbase + kb + 16);
  bf16x16 r;
#pragma unroll
  for (int i = 0; i < 8; ++i) { r[i] = lo[i]; r[i + 8] = hi[i]; }
  return r;
}

// B-fragment (32x16 bf16): lane l -> col n=l&15, k = kb..kb+7 and
// kb+16..kb+23 with kb = (l>=16)*8. Tile stored as rows-of-n (i.e. W[n][k]), ld=32.
__device__ __forceinline__ bf16x16 load_frag_b(const bf16* rowbase, int lane) {
  const int kb = (lane >> 4) << 3;
  bf16x8 lo = *reinterpret_cast<const bf16x8*>(rowbase + kb);
  bf16x8 hi = *reinterpret_cast<const bf16x8*>(rowbase + kb + 16);
  bf16x16 r;
#pragma unroll
  for (int i = 0; i < 8; ++i) { r[i] = lo[i]; r[i + 8] = hi[i]; }
  return r;
}

#define WMMA_BF16(a, b, c) \
  __builtin_amdgcn_wmma_f32_16x16x32_bf16(false, (a), false, (b), (short)0, (c), false, false)

// ---------------------------------------------------------------------------
// LayerNorm (one block per row of 512) -> bf16
// ---------------------------------------------------------------------------
__global__ __launch_bounds__(256)
void layernorm_kernel(const float* __restrict__ x, const float* __restrict__ gamma,
                      const float* __restrict__ beta, bf16* __restrict__ out) {
  const long row = blockIdx.x;
  const float* p = x + row * kD;
  const int t = threadIdx.x;
  float a0 = p[t], a1 = p[t + 256];
  __shared__ float red[256];
  red[t] = a0 + a1;
  __syncthreads();
  for (int s = 128; s > 0; s >>= 1) { if (t < s) red[t] += red[t + s]; __syncthreads(); }
  const float mu = red[0] * (1.0f / (float)kD);
  __syncthreads();
  red[t] = a0 * a0 + a1 * a1;
  __syncthreads();
  for (int s = 128; s > 0; s >>= 1) { if (t < s) red[t] += red[t + s]; __syncthreads(); }
  const float var = red[0] * (1.0f / (float)kD) - mu * mu;
  const float rstd = rsqrtf(var + 1e-5f);
  bf16* o = out + row * kD;
  o[t]       = (bf16)((a0 - mu) * rstd * gamma[t]       + beta[t]);
  o[t + 256] = (bf16)((a1 - mu) * rstd * gamma[t + 256] + beta[t + 256]);
}

// ---------------------------------------------------------------------------
// fp32 -> bf16 convert (weights)
// ---------------------------------------------------------------------------
__global__ __launch_bounds__(256)
void cvt_bf16_kernel(const float* __restrict__ in, bf16* __restrict__ out, long n) {
  long i = (long)blockIdx.x * blockDim.x + threadIdx.x;
  if (i < n) out[i] = (bf16)in[i];
}

// ---------------------------------------------------------------------------
// Sinusoidal positional encoding -> bf16 [S, D]
// ---------------------------------------------------------------------------
__global__ __launch_bounds__(256)
void pe_kernel(bf16* __restrict__ pe) {
  long idx = (long)blockIdx.x * blockDim.x + threadIdx.x;  // over S * D/2
  if (idx >= (long)kS * (kD / 2)) return;
  int s = (int)(idx / (kD / 2));
  int i = (int)(idx % (kD / 2));
  const float c = -9.210340371976184f / (float)kD;  // -ln(10000)/D
  float freq = expf(c * (2.0f * (float)i));
  float ang = (float)s * freq;
  pe[(long)s * kD + 2 * i]     = (bf16)sinf(ang);
  pe[(long)s * kD + 2 * i + 1] = (bf16)cosf(ang);
}

// ---------------------------------------------------------------------------
// Generic WMMA GEMM: C[m,n] = sum_k A[m,k] * W[n,k]  (+ bias)
// Block = 128 threads (4 waves), tile 64x64, K-step 32.
// Double-buffered LDS filled by async global->LDS copies; the fill of step
// i+1 overlaps the WMMAs of step i; s_wait_asynccnt 0 + barrier per step.
// grid.z offsets A/W/out by aZ/wZ/oZ elements (used for per-head slices).
// ---------------------------------------------------------------------------
enum { MODE_BF16 = 0, MODE_QUQV = 1, MODE_F32 = 2, MODE_SCORE = 3 };

template <int MODE>
__global__ __launch_bounds__(128)
void gemm_kernel(const bf16* __restrict__ A, const bf16* __restrict__ W,
                 const float* __restrict__ bias,
                 bf16* __restrict__ outb0, bf16* __restrict__ outb1,
                 float* __restrict__ outf,
                 const float* __restrict__ ps,      // MODE_SCORE: [z][S][S]
                 const float* __restrict__ ubias,   // MODE_QUQV
                 const float* __restrict__ vbias,   // MODE_QUQV
                 int M, int N, int K, int lda, int ldw, int ldo,
                 long aZ, long wZ, long oZ, float scale, int S) {
  __shared__ __align__(32) bf16 As[2][64 * 32];
  __shared__ __align__(32) bf16 Ws[2][64 * 32];
  const int tid  = threadIdx.x;
  const int wv   = tid >> 5;
  const int lane = tid & 31;
  const int n0   = blockIdx.x * 64;
  const int m0   = blockIdx.y * 64;
  const int z    = blockIdx.z;
  A += (long)z * aZ;
  W += (long)z * wZ;

  fx8 acc[4];
#pragma unroll
  for (int i = 0; i < 4; ++i) acc[i] = zero8();

  const int lrow = tid >> 1;          // 0..63
  const int lcol = (tid & 1) * 16;    // 0 or 16
  const bf16* gA = A + (long)(m0 + lrow) * lda + lcol;
  const bf16* gW = W + (long)(n0 + lrow) * ldw + lcol;

  auto stage = [&](int buf, int k0) {
    async_b128(gA + k0,     &As[buf][lrow * 32 + lcol]);
    async_b128(gA + k0 + 8, &As[buf][lrow * 32 + lcol + 8]);
    async_b128(gW + k0,     &Ws[buf][lrow * 32 + lcol]);
    async_b128(gW + k0 + 8, &Ws[buf][lrow * 32 + lcol + 8]);
  };

  stage(0, 0);
  wait_async0();
  __syncthreads();

  const int nk = K / 32;
  for (int i = 0; i < nk; ++i) {
    const int cur = i & 1;
    if (i + 1 < nk) stage(cur ^ 1, (i + 1) * 32);   // overlap fill with WMMAs
    bf16x16 af = load_frag_a(&As[cur][(wv * 16 + (lane & 15)) * 32], lane);
#pragma unroll
    for (int nb = 0; nb < 4; ++nb) {
      bf16x16 bf = load_frag_b(&Ws[cur][(nb * 16 + (lane & 15)) * 32], lane);
      acc[nb] = WMMA_BF16(af, bf, acc[nb]);
    }
    wait_async0();     // own async fill of the other buffer complete
    __syncthreads();   // all waves done reading cur / done filling next
  }

  const int rowoff = (lane >> 4) * 8;
#pragma unroll
  for (int nb = 0; nb < 4; ++nb) {
    const int n = n0 + nb * 16 + (lane & 15);
    const float bn = bias ? bias[n] : 0.0f;
#pragma unroll
    for (int r = 0; r < 8; ++r) {
      const int m = m0 + wv * 16 + rowoff + r;
      float val = acc[nb][r] + bn;
      const long oidx = (long)z * oZ + (long)m * ldo + n;
      if (MODE == MODE_BF16) {
        outb0[oidx] = (bf16)val;
      } else if (MODE == MODE_QUQV) {
        outb0[oidx] = (bf16)(val + ubias[n]);
        outb1[oidx] = (bf16)(val + vbias[n]);
      } else if (MODE == MODE_F32) {
        outf[oidx] = val;
      } else {  // MODE_SCORE: content + rel_shift(pos_score), scaled
        const int i2 = m, j = n;
        float p;
        if (j <= i2)           p = ps[(long)z * oZ + (long)i2 * S + (j + S - 1 - i2)];
        else if (j == i2 + 1)  p = 0.0f;
        else                   p = ps[(long)z * oZ + (long)(i2 + 1) * S + (j - i2 - 2)];
        outf[oidx] = (val + p) * scale;
      }
    }
  }
}

// ---------------------------------------------------------------------------
// Row softmax over S elements, one block per (h, i) row, in place.
// ---------------------------------------------------------------------------
__global__ __launch_bounds__(256)
void softmax_kernel(float* __restrict__ sc, int S) {
  float* p = sc + (long)blockIdx.x * S;
  const int t = threadIdx.x;
  __shared__ float red[256];
  float mx = -1e30f;
  for (int j = t; j < S; j += 256) mx = fmaxf(mx, p[j]);
  red[t] = mx;
  __syncthreads();
  for (int s = 128; s > 0; s >>= 1) { if (t < s) red[t] = fmaxf(red[t], red[t + s]); __syncthreads(); }
  mx = red[0];
  __syncthreads();
  float sum = 0.0f;
  for (int j = t; j < S; j += 256) { float e = expf(p[j] - mx); p[j] = e; sum += e; }
  red[t] = sum;
  __syncthreads();
  for (int s = 128; s > 0; s >>= 1) { if (t < s) red[t] += red[t + s]; __syncthreads(); }
  const float inv = 1.0f / red[0];
  for (int j = t; j < S; j += 256) p[j] *= inv;
}

// ---------------------------------------------------------------------------
// ctx[i, h*64+d] = sum_t attn[h][i][t] * V[t, h*64+d]   (per batch)
// A staged fp32->bf16 into LDS; V tile transposed in LDS so B-fragments are
// contiguous ds_load_b128. grid = (1, S/64, H).
// ---------------------------------------------------------------------------
__global__ __launch_bounds__(128)
void ctx_gemm_kernel(const float* __restrict__ attn, const bf16* __restrict__ V,
                     bf16* __restrict__ ctx, int S) {
  __shared__ __align__(32) bf16 As[64 * 32];
  __shared__ __align__(32) bf16 Vt[64 * 32];  // [d][t]
  const int tid  = threadIdx.x;
  const int wv   = tid >> 5;
  const int lane = tid & 31;
  const int h    = blockIdx.z;
  const int m0   = blockIdx.y * 64;
  const float* Ah = attn + (long)h * S * S;

  fx8 acc[4];
#pragma unroll
  for (int i = 0; i < 4; ++i) acc[i] = zero8();

  const int lrow = tid >> 1;
  const int lcol = (tid & 1) * 16;
  const int tl = tid >> 2;          // 0..31  (t within tile)
  const int dl = (tid & 3) * 16;    // 0,16,32,48 (d base)

  for (int k0 = 0; k0 < S; k0 += 32) {
    // attn tile fp32 -> bf16 LDS (row-major [i][t])
    const float* gp = Ah + (long)(m0 + lrow) * S + k0 + lcol;
#pragma unroll
    for (int q = 0; q < 4; ++q) {
      fx4 f = *reinterpret_cast<const fx4*>(gp + 4 * q);
#pragma unroll
      for (int e = 0; e < 4; ++e) As[lrow * 32 + lcol + 4 * q + e] = (bf16)f[e];
    }
    // V tile transposed into LDS: Vt[d][t]
    const bf16* gv = V + (long)(k0 + tl) * kD + h * 64 + dl;
    bf16x8 v0 = *reinterpret_cast<const bf16x8*>(gv);
    bf16x8 v1 = *reinterpret_cast<const bf16x8*>(gv + 8);
#pragma unroll
    for (int e = 0; e < 8; ++e) {
      Vt[(dl + e) * 32 + tl]     = v0[e];
      Vt[(dl + 8 + e) * 32 + tl] = v1[e];
    }
    __syncthreads();
    bf16x16 af = load_frag_a(&As[(wv * 16 + (lane & 15)) * 32], lane);
#pragma unroll
    for (int nb = 0; nb < 4; ++nb) {
      bf16x16 bf = load_frag_b(&Vt[(nb * 16 + (lane & 15)) * 32], lane);
      acc[nb] = WMMA_BF16(af, bf, acc[nb]);
    }
    __syncthreads();
  }

  const int rowoff = (lane >> 4) * 8;
#pragma unroll
  for (int nb = 0; nb < 4; ++nb) {
    const int d = nb * 16 + (lane & 15);
#pragma unroll
    for (int r = 0; r < 8; ++r) {
      const int m = m0 + wv * 16 + rowoff + r;
      ctx[(long)m * kD + h * 64 + d] = (bf16)acc[nb][r];
    }
  }
}

// ---------------------------------------------------------------------------
extern "C" void kernel_launch(void* const* d_in, const int* in_sizes, int n_in,
                              void* d_out, int out_size, void* d_ws, size_t ws_size,
                              hipStream_t stream) {
  (void)in_sizes; (void)n_in; (void)out_size; (void)ws_size;
  const float* inp   = (const float*)d_in[0];
  const float* gamma = (const float*)d_in[1];
  const float* beta  = (const float*)d_in[2];
  const float* Wq    = (const float*)d_in[3];
  const float* bq    = (const float*)d_in[4];
  const float* Wk    = (const float*)d_in[5];
  const float* bk    = (const float*)d_in[6];
  const float* Wv    = (const float*)d_in[7];
  const float* bv    = (const float*)d_in[8];
  const float* Wpos  = (const float*)d_in[9];
  const float* ubias = (const float*)d_in[10];  // [H*dh] = [512]
  const float* vbias = (const float*)d_in[11];
  const float* Wo    = (const float*)d_in[12];
  const float* bo    = (const float*)d_in[13];

  char* w = (char*)d_ws;
  auto carve = [&](size_t bytes) -> void* {
    void* p = (void*)w;
    w += (bytes + 255) & ~(size_t)255;
    return p;
  };
  const long NT = (long)kB * kS;  // 8192 tokens
  bf16* xb   = (bf16*)carve(NT * kD * 2);
  bf16* qu   = (bf16*)carve(NT * kD * 2);
  bf16* qv   = (bf16*)carve(NT * kD * 2);
  bf16* kbuf = (bf16*)carve(NT * kD * 2);
  bf16* vbuf = (bf16*)carve(NT * kD * 2);
  bf16* ctxb = (bf16*)carve(NT * kD * 2);
  bf16* peb  = (bf16*)carve((long)kS * kD * 2);
  bf16* posb = (bf16*)carve((long)kS * kD * 2);
  bf16* wqb  = (bf16*)carve((long)kD * kD * 2);
  bf16* wkb  = (bf16*)carve((long)kD * kD * 2);
  bf16* wvb  = (bf16*)carve((long)kD * kD * 2);
  bf16* wpb  = (bf16*)carve((long)kD * kD * 2);
  bf16* wob  = (bf16*)carve((long)kD * kD * 2);
  float* psbuf = (float*)carve((long)kH * kS * kS * 4);  // 134 MB, reused per batch
  float* scbuf = (float*)carve((long)kH * kS * kS * 4);  // 134 MB, reused per batch

  const float scale = 0.044194173824159216f;  // 1/sqrt(512)

  // 1) layernorm -> bf16
  layernorm_kernel<<<(int)NT, 256, 0, stream>>>(inp, gamma, beta, xb);

  // 2) weights -> bf16
  const long WN = (long)kD * kD;
  cvt_bf16_kernel<<<(int)((WN + 255) / 256), 256, 0, stream>>>(Wq,   wqb, WN);
  cvt_bf16_kernel<<<(int)((WN + 255) / 256), 256, 0, stream>>>(Wk,   wkb, WN);
  cvt_bf16_kernel<<<(int)((WN + 255) / 256), 256, 0, stream>>>(Wv,   wvb, WN);
  cvt_bf16_kernel<<<(int)((WN + 255) / 256), 256, 0, stream>>>(Wpos, wpb, WN);
  cvt_bf16_kernel<<<(int)((WN + 255) / 256), 256, 0, stream>>>(Wo,   wob, WN);

  // 3) sinusoidal PE
  {
    const long n = (long)kS * (kD / 2);
    pe_kernel<<<(int)((n + 255) / 256), 256, 0, stream>>>(peb);
  }

  // 4) projections (WMMA)
  dim3 blk(128);
  dim3 gProj(kD / 64, (int)(NT / 64), 1);
  gemm_kernel<MODE_QUQV><<<gProj, blk, 0, stream>>>(
      xb, wqb, bq, qu, qv, nullptr, nullptr, ubias, vbias,
      (int)NT, kD, kD, kD, kD, kD, 0L, 0L, 0L, 1.0f, kS);
  gemm_kernel<MODE_BF16><<<gProj, blk, 0, stream>>>(
      xb, wkb, bk, kbuf, nullptr, nullptr, nullptr, nullptr, nullptr,
      (int)NT, kD, kD, kD, kD, kD, 0L, 0L, 0L, 1.0f, kS);
  gemm_kernel<MODE_BF16><<<gProj, blk, 0, stream>>>(
      xb, wvb, bv, vbuf, nullptr, nullptr, nullptr, nullptr, nullptr,
      (int)NT, kD, kD, kD, kD, kD, 0L, 0L, 0L, 1.0f, kS);
  dim3 gPos(kD / 64, kS / 64, 1);
  gemm_kernel<MODE_BF16><<<gPos, blk, 0, stream>>>(
      peb, wpb, nullptr, posb, nullptr, nullptr, nullptr, nullptr, nullptr,
      kS, kD, kD, kD, kD, kD, 0L, 0L, 0L, 1.0f, kS);

  // 5) attention per batch (serialized on stream; reuses psbuf/scbuf)
  const long ZS = (long)kS * kS;
  dim3 gScore(kS / 64, kS / 64, kH);
  dim3 gCtx(1, kS / 64, kH);
  for (int b = 0; b < kB; ++b) {
    const bf16* qvB = qv   + (long)b * kS * kD;
    const bf16* quB = qu   + (long)b * kS * kD;
    const bf16* kB_ = kbuf + (long)b * kS * kD;
    const bf16* vB  = vbuf + (long)b * kS * kD;
    // pos_score[h][i][t] = (q+v_bias) . pos   (per-head K=64 slice via grid.z)
    gemm_kernel<MODE_F32><<<gScore, blk, 0, stream>>>(
        qvB, posb, nullptr, nullptr, nullptr, psbuf, nullptr, nullptr, nullptr,
        kS, kS, 64, kD, kD, kS, 64L, 64L, ZS, 1.0f, kS);
    // score = (content + rel_shift(pos_score)) / sqrt(D)
    gemm_kernel<MODE_SCORE><<<gScore, blk, 0, stream>>>(
        quB, kB_, nullptr, nullptr, nullptr, scbuf, psbuf, nullptr, nullptr,
        kS, kS, 64, kD, kD, kS, 64L, 64L, ZS, scale, kS);
    // softmax over t
    softmax_kernel<<<kH * kS, 256, 0, stream>>>(scbuf, kS);
    // ctx = attn @ V
    ctx_gemm_kernel<<<gCtx, blk, 0, stream>>>(
        scbuf, vB, ctxb + (long)b * kS * kD, kS);
  }

  // 6) out = ctx @ Wo^T + bo  (fp32 output)
  gemm_kernel<MODE_F32><<<gProj, blk, 0, stream>>>(
      ctxb, wob, bo, nullptr, nullptr, (float*)d_out, nullptr, nullptr, nullptr,
      (int)NT, kD, kD, kD, kD, kD, 0L, 0L, 0L, 1.0f, kS);
}
